// SinkAttentionSPWrapper_3779571220969
// MI455X (gfx1250) — matrix operand order repaired and checked by reading, not compiled, not run on hardware
//
#include <hip/hip_runtime.h>
#include <hip/hip_bf16.h>

typedef __attribute__((ext_vector_type(16))) _Float16 v16h;
typedef __attribute__((ext_vector_type(8)))  _Float16 v8h;
typedef __attribute__((ext_vector_type(8)))  float    v8f;

#define NCTX  2048
#define DHEAD 128
#define KP 136   // Klds pitch in halves (272B rows: 16B-aligned, conflict-free reads)
#define VP 40    // Vlds pitch in halves (80B rows: 16B-aligned)
#define PP 40    // Plds pitch in halves

static __device__ __forceinline__ v16h cat8(v8h lo, v8h hi) {
    return __builtin_shufflevector(lo, hi, 0,1,2,3,4,5,6,7,8,9,10,11,12,13,14,15);
}

// Issue global loads for one 32-key K/V tile into registers (no wait here:
// the consumer is next iteration's LDS-store phase, so the loads overlap
// with the current tile's WMMA + softmax work).
static __device__ __forceinline__ void prefetch_tile(const float* __restrict__ kb_,
                                                     const float* __restrict__ vb,
                                                     int jk, int rb, int c4,
                                                     float4* kf, float4* vf) {
#pragma unroll
    for (int e = 0; e < 8; ++e) {
        int r = e * 4 + rb;  // key row within tile (0..31)
        kf[e] = *(const float4*)(kb_ + (size_t)(jk + r) * DHEAD + c4);
        vf[e] = *(const float4*)(vb + (size_t)(jk + r) * DHEAD + c4);
    }
}

__global__ __launch_bounds__(128)
void sink_swa_fattn(const float* __restrict__ qg, const float* __restrict__ kg,
                    const float* __restrict__ vg, const int* __restrict__ nsp,
                    const int* __restrict__ wsp, float* __restrict__ og) {
    __shared__ alignas(16) _Float16 Klds[32 * KP];        // K tile, row-major [key][dim]
    __shared__ alignas(16) _Float16 Vlds[DHEAD * VP];     // V tile, transposed [dim][key]
    __shared__ alignas(16) _Float16 Plds[4][16 * PP];     // per-wave P tile [row][key]

    const int tid  = threadIdx.x;
    const int lane = tid & 31;
    const int wv   = tid >> 5;
    const int n    = lane & 15;   // column / row-in-A index
    const int lg   = lane >> 4;   // lane group (0: lanes 0-15, 1: lanes 16-31)

    const int qtiles = NCTX / 64;
    const int bh = blockIdx.x / qtiles;
    const int q0 = (blockIdx.x % qtiles) * 64;
    const int qw0 = q0 + wv * 16;           // this wave's first query row

    const int ns = nsp[0];
    const int ws = wsp[0];

    const size_t base = (size_t)bh * NCTX * DHEAD;
    const float* qb = qg + base;
    const float* kb_ = kg + base;
    const float* vb = vg + base;
    float* ob = og + base;

    // ---- Load Q tile as 4 A-fragments (16x32 f16), pre-scaled by 1/sqrt(D) ----
    const float scale = 0.08838834764831845f;  // 1/sqrt(128)
    v16h aq[4];
    {
        const float* qr = qb + (size_t)(qw0 + n) * DHEAD;
        const int kbq = lg * 8;
#pragma unroll
        for (int c = 0; c < 4; ++c) {
            const float* p0 = qr + c * 32 + kbq;
#pragma unroll
            for (int h = 0; h < 8; ++h) {
                aq[c][h]     = (_Float16)(p0[h] * scale);       // K = kb + h
                aq[c][h + 8] = (_Float16)(p0[h + 16] * scale);  // K = kb + 16 + h
            }
        }
    }

    // ---- Flash accumulator state ----
    v8f zero = {};
    v8f o[8];
#pragma unroll
    for (int c = 0; c < 8; ++c) o[c] = zero;
    float mrow[8], lrow[8];
#pragma unroll
    for (int r = 0; r < 8; ++r) { mrow[r] = -3.0e38f; lrow[r] = 0.0f; }

    // ---- Key-tile range: sink tile 0 + sliding-window tiles ----
    int jlo = q0 - ws + 1; if (jlo < 0) jlo = 0;
    int tlo = jlo >> 5;
    int thi = (q0 + 63) >> 5;
    int extra = (tlo > 0) ? 1 : 0;
    int nsteps = thi - tlo + 1 + extra;

    const int c4 = (tid & 31) * 4;   // dim quad this thread stages
    const int rb = tid >> 5;         // row sub-index for staging

    const float LOG2E = 1.4426950408889634f;

    // ---- Software pipeline: prefetch first tile (tile 0) into registers ----
    float4 kf[8], vf[8];
    prefetch_tile(kb_, vb, /*jk=*/0, rb, c4, kf, vf);

    for (int step = 0; step < nsteps; ++step) {
        int jk = (extra ? (step == 0 ? 0 : tlo + step - 1) : step) * 32;

        __syncthreads();
        // ---- Drain staged registers into LDS as f16 (V transposed) ----
#pragma unroll
        for (int e = 0; e < 8; ++e) {
            int r = e * 4 + rb;
            _Float16* kd = &Klds[r * KP + c4];
            kd[0] = (_Float16)kf[e].x; kd[1] = (_Float16)kf[e].y;
            kd[2] = (_Float16)kf[e].z; kd[3] = (_Float16)kf[e].w;
            Vlds[(c4 + 0) * VP + r] = (_Float16)vf[e].x;
            Vlds[(c4 + 1) * VP + r] = (_Float16)vf[e].y;
            Vlds[(c4 + 2) * VP + r] = (_Float16)vf[e].z;
            Vlds[(c4 + 3) * VP + r] = (_Float16)vf[e].w;
        }
        __syncthreads();

        // ---- Kick off next tile's global loads; they complete during compute ----
        if (step + 1 < nsteps) {
            int jn = (extra ? tlo + step : step + 1) * 32;
            prefetch_tile(kb_, vb, jn, rb, c4, kf, vf);
        }

        if (jk <= qw0 + 15) {   // tile overlaps this wave's rows
            // ---- S = Q * K^T : two 16-key column groups, 4 K-chunks each ----
            v8f s01[2];
#pragma unroll
            for (int ng = 0; ng < 2; ++ng) {
                v8f s = zero;
#pragma unroll
                for (int c = 0; c < 4; ++c) {
                    const v8h* kp = (const v8h*)&Klds[(ng * 16 + n) * KP + c * 32 + lg * 8];
                    v16h bf = cat8(kp[0], kp[2]);
                    s = __builtin_amdgcn_wmma_f32_16x16x32_f16(false, aq[c], false, bf,
                                                               (short)0, s, false, false);
                }
                s01[ng] = s;
            }

            // ---- Mask + online softmax; write P (f16) to per-wave LDS ----
#pragma unroll
            for (int r = 0; r < 8; ++r) {
                int i  = qw0 + r + 8 * lg;        // query position of this C/D row
                int j0 = jk + n;
                int j1 = j0 + 16;
                float s0 = s01[0][r];
                float s1 = s01[1][r];
                bool a0 = (j0 <= i) && ((j0 < ns) || ((i - j0) < ws));
                bool a1 = (j1 <= i) && ((j1 < ns) || ((i - j1) < ws));
                s0 = a0 ? s0 : -3.0e38f;
                s1 = a1 ? s1 : -3.0e38f;
                float t0 = fmaxf(s0, s1);
#pragma unroll
                for (int msk = 1; msk < 16; msk <<= 1)
                    t0 = fmaxf(t0, __shfl_xor(t0, msk, 32));
                float mnew  = fmaxf(mrow[r], t0);
                float alpha = (mrow[r] <= -1.0e37f) ? 0.0f : exp2f((mrow[r] - mnew) * LOG2E);
                float p0 = (s0 <= -1.0e37f) ? 0.0f : exp2f((s0 - mnew) * LOG2E);
                float p1 = (s1 <= -1.0e37f) ? 0.0f : exp2f((s1 - mnew) * LOG2E);
                float rs = p0 + p1;
#pragma unroll
                for (int msk = 1; msk < 16; msk <<= 1)
                    rs += __shfl_xor(rs, msk, 32);
                lrow[r] = alpha * lrow[r] + rs;
                mrow[r] = mnew;
#pragma unroll
                for (int c = 0; c < 8; ++c) o[c][r] *= alpha;
                int m = r + 8 * lg;
                Plds[wv][m * PP + n]      = (_Float16)p0;
                Plds[wv][m * PP + 16 + n] = (_Float16)p1;
            }

            // ---- O += P * V (one 16x32 A-fragment, 8 output dim-chunks) ----
            {
                const v8h* pp0 = (const v8h*)&Plds[wv][n * PP + lg * 8];
                const v8h* pp1 = (const v8h*)&Plds[wv][n * PP + lg * 8 + 16];
                v16h ap = cat8(pp0[0], pp1[0]);
#pragma unroll
                for (int c = 0; c < 8; ++c) {
                    const v8h* vp = (const v8h*)&Vlds[(c * 16 + n) * VP + lg * 8];
                    v16h bv = cat8(vp[0], vp[2]);
                    o[c] = __builtin_amdgcn_wmma_f32_16x16x32_f16(false, ap, false, bv,
                                                                  (short)0, o[c], false, false);
                }
            }
        }
    }

    // ---- Normalize and write output (fp32) ----
#pragma unroll
    for (int r = 0; r < 8; ++r) {
        float invl = (lrow[r] > 0.0f) ? (1.0f / lrow[r]) : 0.0f;
        float* orow = ob + (size_t)(qw0 + r + 8 * lg) * DHEAD;
#pragma unroll
        for (int c = 0; c < 8; ++c)
            orow[c * 16 + n] = o[c][r] * invl;
    }
}

extern "C" void kernel_launch(void* const* d_in, const int* in_sizes, int n_in,
                              void* d_out, int out_size, void* d_ws, size_t ws_size,
                              hipStream_t stream) {
    (void)in_sizes; (void)n_in; (void)out_size; (void)d_ws; (void)ws_size;
    const float* q = (const float*)d_in[0];
    const float* k = (const float*)d_in[1];
    const float* v = (const float*)d_in[2];
    const int* ns  = (const int*)d_in[3];
    const int* ws  = (const int*)d_in[4];
    float* out = (float*)d_out;

    const int BH = 2 * 16;                 // B*H
    dim3 grid(BH * (NCTX / 64));
    dim3 block(128);                       // 4 wave32 waves
    hipLaunchKernelGGL(sink_swa_fattn, grid, block, 0, stream, q, k, v, ns, ws, out);
}
